// SelfAttention_75393855914584
// MI455X (gfx1250) — matrix operand order repaired and checked by reading, not compiled, not run on hardware
//
#include <hip/hip_runtime.h>
#include <hip/hip_bf16.h>
#include <stdint.h>

// ---------------------------------------------------------------------------
// Problem constants (match the reference)
// ---------------------------------------------------------------------------
#define BSZ   4
#define LSZ   2048
#define DSZ   768
#define INTER 64
#define MRA   64
#define BAND  160          // padded band: keys [q0-64, q0+95] for a 16-row q tile
#define NEGV  1.0e9f

typedef __attribute__((ext_vector_type(16))) __bf16          v16bf;
typedef __attribute__((ext_vector_type(8)))  float           v8f;
typedef __attribute__((ext_vector_type(8)))  unsigned short  u16x8;
typedef __attribute__((ext_vector_type(4)))  float           f32x4;
typedef __attribute__((ext_vector_type(4)))  unsigned short  u16x4;

union ABReg { u16x8 h[2]; v16bf v; };

__device__ __forceinline__ int iclamp(int x, int lo, int hi) {
  return x < lo ? lo : (x > hi ? hi : x);
}

// fp32 -> bf16 round-to-nearest-even
__device__ __forceinline__ unsigned short f2bf(float f) {
  unsigned u = __float_as_uint(f);
  unsigned r = u + 0x7FFFu + ((u >> 16) & 1u);
  return (unsigned short)(r >> 16);
}

__device__ __forceinline__ v8f wmma_bf16(v16bf a, v16bf b, v8f c) {
  // D = A(16x32 bf16) * B(32x16 bf16) + C(16x16 f32)
  return __builtin_amdgcn_wmma_f32_16x16x32_bf16(
      /*neg_a=*/false, a, /*neg_b=*/false, b,
      /*c_mod=*/(short)0, c, /*reuse_a=*/false, /*reuse_b=*/false);
}

// A operand: row-major [rows][ld] bf16; lane covers row, K runs
// {k0+half*8 .. +7} and {k0+16+half*8 .. +7}  (ISA 16-bit A 16x32 layout)
__device__ __forceinline__ v16bf load_a16(const unsigned short* base, int ld,
                                          int row, int k0, int half) {
  ABReg r;
  const unsigned short* p = base + (size_t)row * ld + k0 + half * 8;
  r.h[0] = *(const u16x8*)(p);
  r.h[1] = *(const u16x8*)(p + 16);
  return r.v;
}

// B operand: column n stored as a row of a [N][ld] (i.e. transposed) bf16
// array; lane covers col, K runs {k0+half*8 .. +7} and {k0+16+half*8 .. +7}
__device__ __forceinline__ v16bf load_b16(const unsigned short* base, int ld,
                                          int col, int k0, int half) {
  ABReg r;
  const unsigned short* p = base + (size_t)col * ld + k0 + half * 8;
  r.h[0] = *(const u16x8*)(p);
  r.h[1] = *(const u16x8*)(p + 16);
  return r.v;
}

// Async global -> LDS copy, 16B per lane, tracked with ASYNCcnt (CDNA5 TDM-
// adjacent path; see CDNA5 ISA).  lds_off = LDS byte addr
// (low 32 bits of a generic shared pointer per the aperture mapping).
__device__ __forceinline__ void async_ld_b128(unsigned lds_off, const void* g) {
  asm volatile("global_load_async_to_lds_b128 %0, %1, off"
               :: "v"(lds_off), "v"(g) : "memory");
}
__device__ __forceinline__ void wait_async0() {
  asm volatile("s_wait_asynccnt 0x0" ::: "memory");
}

// ---------------------------------------------------------------------------
// Kernel 1: fp32 -> bf16 convert, 4 elements / thread
// ---------------------------------------------------------------------------
__global__ void cvt_f32_bf16(const float* __restrict__ x,
                             unsigned short* __restrict__ y, int n4) {
  int i = blockIdx.x * blockDim.x + threadIdx.x;
  if (i < n4) {
    f32x4 f = *(const f32x4*)(x + 4 * (size_t)i);
    u16x4 o;
    o.x = f2bf(f.x); o.y = f2bf(f.y); o.z = f2bf(f.z); o.w = f2bf(f.w);
    *(u16x4*)(y + 4 * (size_t)i) = o;
  }
}

// ---------------------------------------------------------------------------
// Kernel 2: W [K][N] fp32  ->  Wt [N][K] bf16   (K = 768)
// ---------------------------------------------------------------------------
__global__ void cvt_w_transpose(const float* __restrict__ w,
                                unsigned short* __restrict__ wt, int K, int N) {
  int i = blockIdx.x * blockDim.x + threadIdx.x;
  if (i < K * N) {
    int n = i / K, k = i % K;
    wt[n * K + k] = f2bf(w[k * N + n]);
  }
}

// ---------------------------------------------------------------------------
// Kernel 3/4: bf16 WMMA GEMM.  A[M][K] bf16 row-major, Bt[N][K] bf16 (= W^T).
// One 16x64 C strip per wave (4 accumulators): A loaded once per K-step and
// reused across 4 WMMAs.  4 waves / block.  N must be a multiple of 64.
// TRANS_OUT=false -> C[M][N] bf16 row-major (Q,K projections)
// TRANS_OUT=true  -> C[N][M] bf16           (V projection, stored transposed)
// ---------------------------------------------------------------------------
template <bool TRANS_OUT>
__global__ void __launch_bounds__(128)
gemm_bf16(const unsigned short* __restrict__ A,
          const unsigned short* __restrict__ Bt,
          unsigned short* __restrict__ C, int M, int N, int K) {
  int wave = threadIdx.x >> 5;
  int lane = threadIdx.x & 31;
  int half = lane >> 4;
  int ln   = lane & 15;

  int strips = N >> 6;                     // 64-wide N strips
  int gw = blockIdx.x * 4 + wave;
  int mt = gw / strips, ns = gw % strips;
  if (mt >= (M >> 4)) return;              // wave-uniform, EXEC stays full
  int m0 = mt << 4, n0 = ns << 6;

  v8f acc[4] = {};
  for (int k0 = 0; k0 < K; k0 += 32) {
    v16bf a = load_a16(A, K, m0 + ln, k0, half);
#pragma unroll
    for (int j = 0; j < 4; ++j) {
      v16bf b = load_b16(Bt, K, n0 + j * 16 + ln, k0, half);
      acc[j] = wmma_bf16(a, b, acc[j]);
    }
  }

#pragma unroll
  for (int j = 0; j < 4; ++j) {
#pragma unroll
    for (int r = 0; r < 8; ++r) {
      int mm = m0 + r + half * 8;          // C layout: M = r + 8*half
      int nn = n0 + j * 16 + ln;           //           N = lane & 15
      unsigned short v = f2bf(acc[j][r]);
      if (TRANS_OUT) C[(size_t)nn * M + mm] = v;
      else           C[(size_t)mm * N + nn] = v;
    }
  }
}

// ---------------------------------------------------------------------------
// Kernel 5: banded attention. One 16-row query tile per block (128 thr/4 wv).
//   Qbf,Kbf : [B*L][64]  bf16
//   Vt      : [768][B*L] bf16 (transposed V projection)
//   out     : [B][L][768] fp32
// ---------------------------------------------------------------------------
__global__ void __launch_bounds__(128)
attn_kernel(const unsigned short* __restrict__ Qbf,
            const unsigned short* __restrict__ Kbf,
            const unsigned short* __restrict__ Vt,
            const float* __restrict__ unk,
            const float* __restrict__ bias,
            float* __restrict__ out) {
  __shared__ __align__(16) unsigned short sK[BAND][INTER];  // staged K band
  __shared__ __align__(16) float          sS[16][BAND];
  __shared__ __align__(16) unsigned short sP[16][BAND];
  __shared__ float sBias[2 * MRA + 1];
  __shared__ float sUnkAdd[BAND];
  __shared__ float sRed[16][8];

  int wave = threadIdx.x >> 5;
  int lane = threadIdx.x & 31;
  int half = lane >> 4;
  int ln   = lane & 15;
  int tid  = threadIdx.x;

  int b       = blockIdx.x / (LSZ / 16);
  int q0      = (blockIdx.x % (LSZ / 16)) << 4;
  int kbase0  = q0 - MRA;                  // first (possibly <0) band key
  int tokBase = b * LSZ;

  // ---- Stage the K-projection band tile into LDS asynchronously ----------
  // 160 rows x 64 bf16 = 20480 B = 1280 x 16B chunks; 10 chunks / thread.
#pragma unroll
  for (int i = 0; i < 10; ++i) {
    int c    = tid + i * 128;              // chunk id
    int row  = c >> 3;                     // band row (8 chunks per 128B row)
    int col8 = (c & 7) * 8;                // bf16 column of chunk
    int tok  = tokBase + iclamp(kbase0 + row, 0, LSZ - 1);
    async_ld_b128((unsigned)(size_t)&sK[row][col8],
                  Kbf + (size_t)tok * INTER + col8);
  }

  // ---- Bias table + additive key mask (sequence bounds + unk) ------------
  if (tid < 2 * MRA + 1) sBias[tid] = bias[tid];
  if (tid < BAND + 128 && tid >= 128) { /* no-op filler to keep EXEC simple */ }
  {
    int idx = tid;                         // 128 threads cover 160 slots in 2
    for (int rep = 0; rep < 2 && idx < BAND; ++rep, idx += 128) {
      int j = kbase0 + idx;
      float add;
      if (j >= 0 && j < LSZ) add = -NEGV * (1.0f - unk[tokBase + j]);
      else                   add = -NEGV;  // off-sequence: force exp -> 0
      sUnkAdd[idx] = add;
    }
  }
  wait_async0();                           // own wave's async chunks done
  __syncthreads();                         // all waves' chunks + tables done

  const float scale = 0.03608439182435161f;  // 1/sqrt(768)

  // ---- Phase 1: S = Q K^T over the band (B operand from LDS) -------------
  v16bf qa0 = load_a16(Qbf, INTER, tokBase + q0 + ln, 0,  half);
  v16bf qa1 = load_a16(Qbf, INTER, tokBase + q0 + ln, 32, half);

  for (int kt = wave; kt < BAND / 16; kt += 4) {
    int bc = kt * 16 + ln;                 // band column handled by this lane
    v8f acc = {};
    acc = wmma_bf16(qa0, load_b16(&sK[0][0], INTER, bc, 0,  half), acc);
    acc = wmma_bf16(qa1, load_b16(&sK[0][0], INTER, bc, 32, half), acc);

#pragma unroll
    for (int r = 0; r < 8; ++r) {
      int m   = r + half * 8;
      int rel = (kbase0 + bc) - (q0 + m);  // key - query
      // branch-free epilogue: scale + clipped bias + additive mask, then
      // band select (single cndmask)
      float s = acc[r] * scale + sBias[iclamp(rel + MRA, 0, 2 * MRA)]
              + sUnkAdd[bc];
      s = (rel >= -MRA && rel <= MRA) ? s : -NEGV;
      sS[m][bc] = s;
    }
  }
  __syncthreads();

  // ---- Phase 2: row softmax, 8 threads per row (20 columns each) ---------
  {
    int row = tid >> 3, seg = tid & 7, c0 = seg * (BAND / 8);
    float mx = -3.0e38f;
    for (int t = 0; t < BAND / 8; ++t) mx = fmaxf(mx, sS[row][c0 + t]);
    sRed[row][seg] = mx;
    __syncthreads();
    mx = fmaxf(fmaxf(fmaxf(sRed[row][0], sRed[row][1]),
                     fmaxf(sRed[row][2], sRed[row][3])),
               fmaxf(fmaxf(sRed[row][4], sRed[row][5]),
                     fmaxf(sRed[row][6], sRed[row][7])));
    float sum = 0.0f;
    for (int t = 0; t < BAND / 8; ++t) sum += __expf(sS[row][c0 + t] - mx);
    __syncthreads();                       // done reading sRed maxima
    sRed[row][seg] = sum;
    __syncthreads();
    sum = (sRed[row][0] + sRed[row][1]) + (sRed[row][2] + sRed[row][3])
        + (sRed[row][4] + sRed[row][5]) + (sRed[row][6] + sRed[row][7]);
    float inv = 1.0f / sum;
    for (int t = 0; t < BAND / 8; ++t)
      sP[row][c0 + t] = f2bf(__expf(sS[row][c0 + t] - mx) * inv);
  }
  __syncthreads();

  // ---- Phase 3: O = P * V  (K-loop over the 160-wide band) ---------------
  for (int nt = wave; nt < DSZ / 16; nt += 4) {
    int n0 = nt << 4;
    v8f acc = {};
#pragma unroll
    for (int kc = 0; kc < BAND / 32; ++kc) {
      // A operand from LDS P (row-major 16 x BAND)
      ABReg ar;
      const unsigned short* pa = &sP[ln][kc * 32 + half * 16];
      ar.h[0] = *(const u16x8*)pa;
      ar.h[1] = *(const u16x8*)(pa + 8);
      // B operand from Vt: column = output dim, K run = 16 consecutive keys.
      // Runs are 16-aligned vs the sequence so they never straddle [0,L);
      // clamped runs correspond to P==0 rows and contribute nothing.
      int krun = iclamp(kbase0 + kc * 32 + half * 16, 0, LSZ - 16);
      const unsigned short* pb =
          Vt + (size_t)(n0 + ln) * (BSZ * LSZ) + tokBase + krun;
      ABReg br;
      br.h[0] = *(const u16x8*)pb;
      br.h[1] = *(const u16x8*)(pb + 8);
      acc = wmma_bf16(ar.v, br.v, acc);
    }
#pragma unroll
    for (int r = 0; r < 8; ++r) {
      int m = r + half * 8;
      out[(size_t)(tokBase + q0 + m) * DSZ + n0 + ln] = acc[r];
    }
  }
}

// ---------------------------------------------------------------------------
// Host launcher
// ---------------------------------------------------------------------------
extern "C" void kernel_launch(void* const* d_in, const int* in_sizes, int n_in,
                              void* d_out, int out_size, void* d_ws, size_t ws_size,
                              hipStream_t stream) {
  (void)in_sizes; (void)n_in; (void)out_size; (void)ws_size;

  const float* inputs = (const float*)d_in[0];   // [B,L,D]
  const float* unk    = (const float*)d_in[1];   // [B,L]
  const float* qw     = (const float*)d_in[2];   // [D,INTER]
  const float* kw     = (const float*)d_in[3];   // [D,INTER]
  const float* vw     = (const float*)d_in[4];   // [D,D]
  const float* bias   = (const float*)d_in[5];   // [2*MRA+1]
  float* out = (float*)d_out;                    // [B,L,D]

  const int NTOK = BSZ * LSZ;                    // 8192 tokens

  // workspace carve-out (bf16 buffers), 256B aligned
  char* ws = (char*)d_ws;
  size_t off = 0;
  auto carve = [&](size_t bytes) {
    char* p = ws + off;
    off = (off + bytes + 255) & ~(size_t)255;
    return (unsigned short*)p;
  };
  unsigned short* Xbf = carve((size_t)NTOK * DSZ * 2);   // 12.6 MB
  unsigned short* Wqt = carve((size_t)INTER * DSZ * 2);
  unsigned short* Wkt = carve((size_t)INTER * DSZ * 2);
  unsigned short* Wvt = carve((size_t)DSZ * DSZ * 2);    // 1.2 MB
  unsigned short* Qbf = carve((size_t)NTOK * INTER * 2); // 1 MB
  unsigned short* Kbf = carve((size_t)NTOK * INTER * 2); // 1 MB
  unsigned short* Vt  = carve((size_t)DSZ * NTOK * 2);   // 12.6 MB

  // 1) convert activations to bf16 (4-wide)
  {
    int n4 = (NTOK * DSZ) / 4;
    cvt_f32_bf16<<<(n4 + 255) / 256, 256, 0, stream>>>(inputs, Xbf, n4);
  }
  // 2) transpose+convert weights
  cvt_w_transpose<<<(DSZ * INTER + 255) / 256, 256, 0, stream>>>(qw, Wqt, DSZ, INTER);
  cvt_w_transpose<<<(DSZ * INTER + 255) / 256, 256, 0, stream>>>(kw, Wkt, DSZ, INTER);
  cvt_w_transpose<<<(DSZ * DSZ   + 255) / 256, 256, 0, stream>>>(vw, Wvt, DSZ, DSZ);

  // 3) Q, K projections: [8192x768] x [768x64] -> row-major bf16
  {
    int waves = (NTOK / 16) * (INTER / 64);     // one 16x64 strip per wave
    gemm_bf16<false><<<waves / 4, 128, 0, stream>>>(Xbf, Wqt, Qbf, NTOK, INTER, DSZ);
    gemm_bf16<false><<<waves / 4, 128, 0, stream>>>(Xbf, Wkt, Kbf, NTOK, INTER, DSZ);
  }
  // 4) V projection: [8192x768] x [768x768] -> transposed bf16 [768][8192]
  {
    int waves = (NTOK / 16) * (DSZ / 64);       // 6144 strips
    gemm_bf16<true><<<waves / 4, 128, 0, stream>>>(Xbf, Wvt, Vt, NTOK, DSZ, DSZ);
  }
  // 5) banded attention, one 16-row query tile per block
  attn_kernel<<<BSZ * (LSZ / 16), 128, 0, stream>>>(Qbf, Kbf, Vt, unk, bias, out);
}
